// MyDecoder_43181601194940
// MI455X (gfx1250) — compile-verified, matrix operand order checked by reading, not run on hardware
//
#include <hip/hip_runtime.h>
#include <math.h>

// Problem constants (match reference)
#define B_   2
#define T_   2048
#define V_   32000
#define D_   1024
#define DF_  4096
#define L_   6
#define BT_  (B_ * T_)
#define EPS_ 1e-5f

typedef __bf16 bf16;
typedef __bf16 v16bf __attribute__((ext_vector_type(16)));
typedef float  v8f   __attribute__((ext_vector_type(8)));

#define AS1 __attribute__((address_space(1)))
#define AS3 __attribute__((address_space(3)))

// CDNA5 async global->LDS staging (ASYNCcnt-tracked), with portable fallback.
#if __has_builtin(__builtin_amdgcn_global_load_async_to_lds_b128)
#define HAVE_ASYNC_LDS 1
#else
#define HAVE_ASYNC_LDS 0
#endif

typedef __attribute__((__vector_size__(4 * sizeof(int)))) int v4i_t;

__device__ __forceinline__ void stage_b128(const bf16* g, bf16* l) {
#if HAVE_ASYNC_LDS
    // global_load_async_to_lds_b128: 16B global -> LDS, no VGPR round-trip
    AS1 v4i_t* gp = (AS1 v4i_t*)(v4i_t*)(void*)const_cast<bf16*>(g);
    AS3 v4i_t* lp = (AS3 v4i_t*)(v4i_t*)(void*)l;
    __builtin_amdgcn_global_load_async_to_lds_b128(gp, lp, 0, 0);
#else
    *(uint4*)l = *(const uint4*)g;
#endif
}

__device__ __forceinline__ void wait_stage() {
#if HAVE_ASYNC_LDS
#if __has_builtin(__builtin_amdgcn_s_wait_asynccnt)
    __builtin_amdgcn_s_wait_asynccnt(0);
#else
    asm volatile("s_wait_asynccnt 0" ::: "memory");
#endif
#endif
}

// ---------------------------------------------------------------------------
// fp32 -> bf16 conversion (grid-stride)
// ---------------------------------------------------------------------------
__global__ __launch_bounds__(256)
void f32_to_bf16_kernel(const float* __restrict__ in, bf16* __restrict__ out, size_t n) {
    size_t i = (size_t)blockIdx.x * blockDim.x + threadIdx.x;
    size_t stride = (size_t)gridDim.x * blockDim.x;
    for (; i < n; i += stride) out[i] = (bf16)in[i];
}

// ---------------------------------------------------------------------------
// Embedding lookup + sinusoidal positional encoding: x[b,t,:] = E[tok] + PE(t)
// ---------------------------------------------------------------------------
__global__ __launch_bounds__(256)
void embed_pe_kernel(const int* __restrict__ tokens, const float* __restrict__ embedW,
                     float* __restrict__ x) {
    int bt = blockIdx.x;             // 0 .. BT-1
    int t  = bt % T_;
    int tok = tokens[bt];
    const float* e = embedW + (size_t)tok * D_;
    float* xr = x + (size_t)bt * D_;
    for (int d = threadIdx.x; d < D_; d += blockDim.x) {
        float expo  = (float)((d >> 1) << 1) / (float)D_;
        float angle = (float)t * powf(10000.0f, -expo);
        float pe    = (d & 1) ? cosf(angle) : sinf(angle);
        xr[d] = e[d] + pe;
    }
}

// ---------------------------------------------------------------------------
// WMMA bf16 GEMM:  C[z] = alpha * A[z] @ op(B[z]) + bias  (opt. ReLU)
//   A: [M,K] bf16 row-major (lda)
//   B: TRANSB=false -> [K,N] row-major (ldb=N);  true -> [N,K] row-major (ldb=K)
//   C: [M,N] fp32 (ldc)
// Block: 256 threads = 8 waves; block tile 64x128; wave tile 32x32
// (4 x v_wmma_f32_16x16x32_bf16 per K-step: 2 A-frags x 2 B-frags).
// Tiles staged to LDS via async global->LDS loads when available.
// Dims assumed: M % 64 == 0, N % 128 == 0, K % 32 == 0 (true for this model).
// ---------------------------------------------------------------------------
template <bool TRANSB>
__global__ __launch_bounds__(256)
void gemm_bf16_wmma(const bf16* __restrict__ A, const bf16* __restrict__ Bm,
                    const float* __restrict__ bias, float* __restrict__ C,
                    int M, int N, int K, int lda, int ldb, int ldc,
                    float alpha, int relu,
                    size_t strideA, size_t strideB, size_t strideC) {
    __shared__ __align__(16) bf16 As[64 * 40];   // 64 rows x 32, stride 40 (16B-aligned rows)
    __shared__ __align__(16) bf16 Bs[5120];      // NN: [32][136]; NT: [128][40]

    const int tid  = threadIdx.x;
    const int lane = tid & 31;
    const int w    = tid >> 5;
    const int wm   = w & 1;        // 2 waves along M
    const int wn   = w >> 1;       // 4 waves along N
    const int half = lane >> 4;    // lane group (K-offset select)
    const int ml   = lane & 15;

    const int z      = blockIdx.z;
    const int blockM = blockIdx.y * 64;
    const int blockN = blockIdx.x * 128;
    const bf16* Ab = A  + strideA * z;
    const bf16* Bb = Bm + strideB * z;
    float*      Cb = C  + strideC * z;

    v8f c00 = {}; v8f c01 = {}; v8f c10 = {}; v8f c11 = {};

    const int ar  = tid >> 2;          // A tile: 64 rows x 32 cols = 256 x 16B
    const int acg = (tid & 3) * 8;

    for (int k0 = 0; k0 < K; k0 += 32) {
        // --- stage A tile (64x32 bf16) ---
        stage_b128(Ab + (size_t)(blockM + ar) * lda + k0 + acg, &As[ar * 40 + acg]);
        // --- stage B tile (512 x 16B, 2 per thread) ---
        if constexpr (TRANSB) {
            #pragma unroll
            for (int j = 0; j < 2; ++j) {
                int lin = tid + j * 256;
                int n = lin >> 2, cg = (lin & 3) * 8;     // 128 rows(n) x 32 cols(k)
                stage_b128(Bb + (size_t)(blockN + n) * ldb + k0 + cg, &Bs[n * 40 + cg]);
            }
        } else {
            #pragma unroll
            for (int j = 0; j < 2; ++j) {
                int lin = tid + j * 256;
                int r = lin >> 4, cg = (lin & 15) * 8;    // 32 rows(k) x 128 cols(n)
                stage_b128(Bb + (size_t)(k0 + r) * ldb + blockN + cg, &Bs[r * 136 + cg]);
            }
        }
        if (k0 + 32 < K)  // prefetch next A K-slab -> global_prefetch_b8
            __builtin_prefetch(Ab + (size_t)(blockM + ar) * lda + (k0 + 32) + acg, 0, 0);
        wait_stage();
        __syncthreads();

        // --- A fragments (ISA 16-bit A layout: K groups by lane half) ---
        const int rA0  = wm * 32 + ml;
        const int rA1  = rA0 + 16;
        const int koff = half * 8;
        v16bf a0, a1;
        #pragma unroll
        for (int i = 0; i < 8; ++i) {
            a0[i]     = As[rA0 * 40 + koff + i];
            a0[i + 8] = As[rA0 * 40 + 16 + koff + i];
            a1[i]     = As[rA1 * 40 + koff + i];
            a1[i + 8] = As[rA1 * 40 + 16 + koff + i];
        }
        // --- B fragments: per-lane column n (B loaded column-major) ---
        const int n0 = wn * 32 + ml;
        const int n1 = n0 + 16;
        v16bf b0, b1;
        if constexpr (TRANSB) {
            #pragma unroll
            for (int i = 0; i < 8; ++i) {
                b0[i]     = Bs[n0 * 40 + koff + i];
                b0[i + 8] = Bs[n0 * 40 + 16 + koff + i];
                b1[i]     = Bs[n1 * 40 + koff + i];
                b1[i + 8] = Bs[n1 * 40 + 16 + koff + i];
            }
        } else {
            #pragma unroll
            for (int i = 0; i < 8; ++i) {
                b0[i]     = Bs[(koff + i) * 136 + n0];
                b0[i + 8] = Bs[(16 + koff + i) * 136 + n0];
                b1[i]     = Bs[(koff + i) * 136 + n1];
                b1[i + 8] = Bs[(16 + koff + i) * 136 + n1];
            }
        }
        c00 = __builtin_amdgcn_wmma_f32_16x16x32_bf16(false, a0, false, b0, (short)0, c00, false, false);
        c01 = __builtin_amdgcn_wmma_f32_16x16x32_bf16(false, a0, false, b1, (short)0, c01, false, false);
        c10 = __builtin_amdgcn_wmma_f32_16x16x32_bf16(false, a1, false, b0, (short)0, c10, false, false);
        c11 = __builtin_amdgcn_wmma_f32_16x16x32_bf16(false, a1, false, b1, (short)0, c11, false, false);
        __syncthreads();
    }

    // --- epilogue per ISA C/D layout: VGPR i -> (M = 8*half + i, N = ml) ---
    #pragma unroll
    for (int i = 0; i < 8; ++i) {
        int gm0 = blockM + wm * 32 + half * 8 + i;
        int gm1 = gm0 + 16;
        int gn0 = blockN + wn * 32 + ml;
        int gn1 = gn0 + 16;
        float v00 = c00[i] * alpha;
        float v01 = c01[i] * alpha;
        float v10 = c10[i] * alpha;
        float v11 = c11[i] * alpha;
        if (bias) {
            float bb0 = bias[gn0], bb1 = bias[gn1];
            v00 += bb0; v01 += bb1; v10 += bb0; v11 += bb1;
        }
        if (relu) {
            v00 = fmaxf(v00, 0.0f); v01 = fmaxf(v01, 0.0f);
            v10 = fmaxf(v10, 0.0f); v11 = fmaxf(v11, 0.0f);
        }
        Cb[(size_t)gm0 * ldc + gn0] = v00;
        Cb[(size_t)gm0 * ldc + gn1] = v01;
        Cb[(size_t)gm1 * ldc + gn0] = v10;
        Cb[(size_t)gm1 * ldc + gn1] = v11;
    }
}

// ---------------------------------------------------------------------------
// Causal softmax over score rows; writes bf16 probs (0 beyond the diagonal)
// One block per (b,t) row.
// ---------------------------------------------------------------------------
__global__ __launch_bounds__(256)
void softmax_causal_kernel(const float* __restrict__ scores, bf16* __restrict__ probs) {
    __shared__ float red[256];
    int row = blockIdx.x;                 // b*T + t
    int t   = row % T_;
    const float* s = scores + (size_t)row * T_;
    bf16*        p = probs  + (size_t)row * T_;
    int tid = threadIdx.x;
    int len = t + 1;

    float m = -3.4e38f;
    for (int i = tid; i < len; i += 256) m = fmaxf(m, s[i]);
    red[tid] = m; __syncthreads();
    for (int st = 128; st > 0; st >>= 1) { if (tid < st) red[tid] = fmaxf(red[tid], red[tid + st]); __syncthreads(); }
    m = red[0]; __syncthreads();

    float sum = 0.0f;
    for (int i = tid; i < len; i += 256) sum += expf(s[i] - m);
    red[tid] = sum; __syncthreads();
    for (int st = 128; st > 0; st >>= 1) { if (tid < st) red[tid] += red[tid + st]; __syncthreads(); }
    float inv = 1.0f / red[0];

    for (int i = tid; i < T_; i += 256) {
        float v = (i < len) ? expf(s[i] - m) * inv : 0.0f;
        p[i] = (bf16)v;
    }
}

// ---------------------------------------------------------------------------
// x = LayerNorm(x + r) * g + beta  (in place on x). One block per row, D=1024.
// ---------------------------------------------------------------------------
__global__ __launch_bounds__(256)
void add_layernorm_kernel(float* __restrict__ x, const float* __restrict__ r,
                          const float* __restrict__ g, const float* __restrict__ beta) {
    __shared__ float red[256];
    int row = blockIdx.x;
    float*       xr = x + (size_t)row * D_;
    const float* rr = r + (size_t)row * D_;
    int tid = threadIdx.x;

    float vals[4];
    float sum = 0.0f;
    #pragma unroll
    for (int j = 0; j < 4; ++j) { int d = tid + j * 256; vals[j] = xr[d] + rr[d]; sum += vals[j]; }
    red[tid] = sum; __syncthreads();
    for (int st = 128; st > 0; st >>= 1) { if (tid < st) red[tid] += red[tid + st]; __syncthreads(); }
    float mu = red[0] * (1.0f / D_); __syncthreads();

    float var = 0.0f;
    #pragma unroll
    for (int j = 0; j < 4; ++j) { float dv = vals[j] - mu; var += dv * dv; }
    red[tid] = var; __syncthreads();
    for (int st = 128; st > 0; st >>= 1) { if (tid < st) red[tid] += red[tid + st]; __syncthreads(); }
    float rs = rsqrtf(red[0] * (1.0f / D_) + EPS_);

    #pragma unroll
    for (int j = 0; j < 4; ++j) {
        int d = tid + j * 256;
        xr[d] = (vals[j] - mu) * rs * g[d] + beta[d];
    }
}

// ---------------------------------------------------------------------------
// In-place log-softmax across V per row. One block per row.
// ---------------------------------------------------------------------------
__global__ __launch_bounds__(256)
void log_softmax_kernel(float* __restrict__ out) {
    __shared__ float red[256];
    int row = blockIdx.x;
    float* o = out + (size_t)row * V_;
    int tid = threadIdx.x;

    float m = -3.4e38f;
    for (int i = tid; i < V_; i += 256) m = fmaxf(m, o[i]);
    red[tid] = m; __syncthreads();
    for (int st = 128; st > 0; st >>= 1) { if (tid < st) red[tid] = fmaxf(red[tid], red[tid + st]); __syncthreads(); }
    m = red[0]; __syncthreads();

    float sum = 0.0f;
    for (int i = tid; i < V_; i += 256) sum += expf(o[i] - m);
    red[tid] = sum; __syncthreads();
    for (int st = 128; st > 0; st >>= 1) { if (tid < st) red[tid] += red[tid + st]; __syncthreads(); }
    float lse = m + logf(red[0]);

    for (int i = tid; i < V_; i += 256) o[i] = o[i] - lse;
}

// ---------------------------------------------------------------------------
// Host orchestration
// ---------------------------------------------------------------------------
extern "C" void kernel_launch(void* const* d_in, const int* in_sizes, int n_in,
                              void* d_out, int out_size, void* d_ws, size_t ws_size,
                              hipStream_t stream) {
    (void)in_sizes; (void)n_in; (void)out_size; (void)ws_size;

    const int*   tokens = (const int*)  d_in[0];
    const float* embedW = (const float*)d_in[1];
    const float* Wq    = (const float*)d_in[2];
    const float* bq    = (const float*)d_in[3];
    const float* Wk    = (const float*)d_in[4];
    const float* bk    = (const float*)d_in[5];
    const float* Wv    = (const float*)d_in[6];
    const float* bv    = (const float*)d_in[7];
    const float* g1    = (const float*)d_in[8];
    const float* beta1 = (const float*)d_in[9];
    const float* W1    = (const float*)d_in[10];
    const float* bf1   = (const float*)d_in[11];
    const float* W2    = (const float*)d_in[12];
    const float* bf2   = (const float*)d_in[13];
    const float* g2    = (const float*)d_in[14];
    const float* beta2 = (const float*)d_in[15];
    const float* Wout  = (const float*)d_in[16];
    const float* bout  = (const float*)d_in[17];
    float* out = (float*)d_out;

    // Workspace carve-out (256B aligned slabs)
    char* wsp = (char*)d_ws;
    auto alloc = [&](size_t bytes) -> void* {
        void* p = wsp; wsp += (bytes + 255) & ~((size_t)255); return p;
    };
    float* x      = (float*)alloc((size_t)BT_ * D_  * 4);
    bf16*  xb     = (bf16*) alloc((size_t)BT_ * D_  * 2);
    float* qf     = (float*)alloc((size_t)BT_ * D_  * 4);
    float* kf     = (float*)alloc((size_t)BT_ * D_  * 4);
    float* vf     = (float*)alloc((size_t)BT_ * D_  * 4);
    bf16*  qb     = (bf16*) alloc((size_t)BT_ * D_  * 2);
    bf16*  kb     = (bf16*) alloc((size_t)BT_ * D_  * 2);
    bf16*  vb     = (bf16*) alloc((size_t)BT_ * D_  * 2);
    float* scores = (float*)alloc((size_t)B_ * T_ * T_ * 4);
    bf16*  probs  = (bf16*) alloc((size_t)B_ * T_ * T_ * 2);
    float* attn   = (float*)alloc((size_t)BT_ * D_  * 4);   // also reused as FFN output
    float* h      = (float*)alloc((size_t)BT_ * DF_ * 4);
    bf16*  hb     = (bf16*) alloc((size_t)BT_ * DF_ * 2);
    bf16*  wb     = (bf16*) alloc((size_t)D_  * V_  * 2);   // largest weight (Wout)

    auto conv = [&](const float* src, bf16* dst, size_t n) {
        int blocks = (int)((n + 2047) / 2048);
        if (blocks > 65535) blocks = 65535;
        f32_to_bf16_kernel<<<blocks, 256, 0, stream>>>(src, dst, n);
    };
    auto gemm = [&](const bf16* A, const bf16* Bm, const float* bias, float* C,
                    int M, int N, int K, int lda, int ldb, int ldc,
                    float alpha, int relu, int batch,
                    size_t sA, size_t sB, size_t sC, bool transb) {
        dim3 grid(N / 128, M / 64, batch);
        if (transb)
            gemm_bf16_wmma<true ><<<grid, 256, 0, stream>>>(A, Bm, bias, C, M, N, K, lda, ldb, ldc, alpha, relu, sA, sB, sC);
        else
            gemm_bf16_wmma<false><<<grid, 256, 0, stream>>>(A, Bm, bias, C, M, N, K, lda, ldb, ldc, alpha, relu, sA, sB, sC);
    };

    // 1) embed + positional encoding
    embed_pe_kernel<<<BT_, 256, 0, stream>>>(tokens, embedW, x);

    const float inv_scale = 1.0f / sqrtf((float)D_);

    for (int l = 0; l < L_; ++l) {
        conv(x, xb, (size_t)BT_ * D_);

        // Q, K, V projections
        conv(Wq + (size_t)l * D_ * D_, wb, (size_t)D_ * D_);
        gemm(xb, wb, bq + (size_t)l * D_, qf, BT_, D_, D_, D_, D_, D_, 1.0f, 0, 1, 0, 0, 0, false);
        conv(Wk + (size_t)l * D_ * D_, wb, (size_t)D_ * D_);
        gemm(xb, wb, bk + (size_t)l * D_, kf, BT_, D_, D_, D_, D_, D_, 1.0f, 0, 1, 0, 0, 0, false);
        conv(Wv + (size_t)l * D_ * D_, wb, (size_t)D_ * D_);
        gemm(xb, wb, bv + (size_t)l * D_, vf, BT_, D_, D_, D_, D_, D_, 1.0f, 0, 1, 0, 0, 0, false);

        conv(qf, qb, (size_t)BT_ * D_);
        conv(kf, kb, (size_t)BT_ * D_);
        conv(vf, vb, (size_t)BT_ * D_);

        // scores[b] = (Q[b] @ K[b]^T) / sqrt(D)   (batched NT GEMM)
        gemm(qb, kb, nullptr, scores, T_, T_, D_, D_, D_, T_,
             inv_scale, 0, B_, (size_t)T_ * D_, (size_t)T_ * D_, (size_t)T_ * T_, true);

        // causal softmax -> bf16 probs
        softmax_causal_kernel<<<BT_, 256, 0, stream>>>(scores, probs);

        // attn[b] = probs[b] @ V[b]   (batched NN GEMM)
        gemm(probs, vb, nullptr, attn, T_, D_, T_, T_, D_, D_,
             1.0f, 0, B_, (size_t)T_ * T_, (size_t)T_ * D_, (size_t)T_ * D_, false);

        // x = LN(x + attn)
        add_layernorm_kernel<<<BT_, 256, 0, stream>>>(x, attn, g1 + (size_t)l * D_, beta1 + (size_t)l * D_);

        // FFN
        conv(x, xb, (size_t)BT_ * D_);
        conv(W1 + (size_t)l * D_ * DF_, wb, (size_t)D_ * DF_);
        gemm(xb, wb, bf1 + (size_t)l * DF_, h, BT_, DF_, D_, D_, DF_, DF_, 1.0f, 1, 1, 0, 0, 0, false);
        conv(h, hb, (size_t)BT_ * DF_);
        conv(W2 + (size_t)l * DF_ * D_, wb, (size_t)DF_ * D_);
        gemm(hb, wb, bf2 + (size_t)l * D_, attn, BT_, D_, DF_, DF_, D_, D_, 1.0f, 0, 1, 0, 0, 0, false);

        // x = LN(x + ffn)
        add_layernorm_kernel<<<BT_, 256, 0, stream>>>(x, attn, g2 + (size_t)l * D_, beta2 + (size_t)l * D_);
    }

    // LM head: logits -> d_out, then in-place log-softmax
    conv(x, xb, (size_t)BT_ * D_);
    conv(Wout, wb, (size_t)D_ * V_);
    gemm(xb, wb, bout, out, BT_, V_, D_, D_, V_, V_, 1.0f, 0, 1, 0, 0, 0, false);
    log_softmax_kernel<<<BT_, 256, 0, stream>>>(out);
}
